// MultiheadAttention_65996467470777
// MI455X (gfx1250) — matrix operand order repaired and checked by reading, not compiled, not run on hardware
//
#include <hip/hip_runtime.h>
#include <hip/hip_bf16.h>

typedef __attribute__((ext_vector_type(16))) _Float16 v16h;
typedef __attribute__((ext_vector_type(8)))  _Float16 v8h;
typedef __attribute__((ext_vector_type(4)))  _Float16 v4h;
typedef __attribute__((ext_vector_type(8)))  float    v8f;

static constexpr int Bsz   = 4;
static constexpr int Lseq  = 1024;
static constexpr int Dm    = 1024;
static constexpr int DH    = 64;          // head dim
static constexpr int Hh    = 16;
static constexpr int Mrows = Bsz * Lseq;  // 4096
static constexpr float NEG_MASK = -4294967295.0f; // -(2^32)+1, matches reference

#define WMMA16x16x32(a, b, c) \
    __builtin_amdgcn_wmma_f32_16x16x32_f16(false, (a), false, (b), (short)0, (c), false, false)

// ---------------------------------------------------------------------------
// Async global->LDS (CDNA5 GLOBAL_LOAD_ASYNC_TO_LDS_B128, ASYNCcnt-tracked).
// Builtin signature (from hipcc diagnostic): arg0 = int4(gcc-vector) AS1 ptr,
// arg1 = int4 AS3 ptr, arg2/3 = imm offset / cpol.
// Guarded: falls back to a synchronous b128 copy if the builtin is absent.
// ---------------------------------------------------------------------------
#if __has_builtin(__builtin_amdgcn_global_load_async_to_lds_b128)
#define ASYNC_LDS 1
typedef int v4i_gcc __attribute__((vector_size(16)));
typedef __attribute__((address_space(1))) v4i_gcc* v4i_gptr;
typedef __attribute__((address_space(3))) v4i_gcc* v4i_lptr;
#else
#define ASYNC_LDS 0
#endif

__device__ __forceinline__ void copy_b128_to_lds(const void* gsrc, void* ldst) {
#if ASYNC_LDS
    // Generic LDS pointers carry the LDS byte offset in their low 32 bits
    // (aperture mapping); global generic pointers are numerically AS1.
    __builtin_amdgcn_global_load_async_to_lds_b128(
        (v4i_gptr)(unsigned long long)gsrc,
        (v4i_lptr)(unsigned int)(unsigned long long)ldst, 0, 0);
#else
    *(v8h*)ldst = *(const v8h*)gsrc;
#endif
}

__device__ __forceinline__ void wait_async_lds() {
#if ASYNC_LDS
#if __has_builtin(__builtin_amdgcn_s_wait_asynccnt)
    __builtin_amdgcn_s_wait_asynccnt(0);
#else
    asm volatile("s_wait_asynccnt 0x0" ::: "memory");
#endif
#endif
}

// ---------------------------------------------------------------------------
// Fragment loaders. All gathers are CONTIGUOUS 16-byte chunks per lane so the
// backend emits ds_load_b128.
// ---------------------------------------------------------------------------

__device__ __forceinline__ v16h join16(v8h lo, v8h hi) {
    v16h r;
#pragma unroll
    for (int i = 0; i < 8; ++i) { r[i] = lo[i]; r[8 + i] = hi[i]; }
    return r;
}

// A-matrix 16x32 f16 (tile stored row-major [M][K], 16B-aligned rows):
// lane L(<16) row M=L holds K 0..7 / 16..23; lane L+16 holds K 8..15 / 24..31.
__device__ __forceinline__ v16h load_a_frag(const _Float16* lds, int row0, int col0,
                                            int ld, int lane) {
    const int hlf = lane >> 4;
    const int r   = lane & 15;
    const _Float16* base = lds + (size_t)(row0 + r) * ld + col0 + hlf * 8;
    const v8h lo = *(const v8h*)(base);       // K  0..7  (or 8..15)
    const v8h hi = *(const v8h*)(base + 16);  // K 16..23 (or 24..31)
    return join16(lo, hi);
}

// B-matrix 32x16 f16 where the source tile is stored TRANSPOSED [N][K]:
// lane column N = lane%16; lanes 0-15 hold K 0..7 / 16..23, lanes 16-31 K 8..15 / 24..31.
__device__ __forceinline__ v16h load_b_frag_T(const _Float16* lds, int k0, int n0,
                                              int ld, int lane) {
    const int hlf = lane >> 4;
    const int n   = lane & 15;
    const _Float16* base = lds + (size_t)(n0 + n) * ld + k0 + hlf * 8;
    const v8h lo = *(const v8h*)(base);
    const v8h hi = *(const v8h*)(base + 16);
    return join16(lo, hi);
}

__device__ __forceinline__ v4h cvt4(float4 v) {
    v4h h;
    h[0] = (_Float16)v.x; h[1] = (_Float16)v.y;
    h[2] = (_Float16)v.z; h[3] = (_Float16)v.w;
    return h;
}

// ---------------------------------------------------------------------------
// GEMM: C[M,N] = act(A[M,K] @ W[N,K]^T + bias)   (torch Linear semantics)
// Block tile 128x64, K-step 32, 8 waves in a 4x2 grid (32x32 C per wave).
// f16 A tiles stream through the async global->LDS path; f32 tiles convert
// through VGPRs (b128 global loads, b64 LDS stores).
// ---------------------------------------------------------------------------
template <bool A_F16, bool RELU, bool OUT_F16>
__global__ __launch_bounds__(256) void gemm_bias_act(const void* __restrict__ Ap,
                                                     const float* __restrict__ W,
                                                     const float* __restrict__ bias,
                                                     void* __restrict__ Cp,
                                                     int M, int N, int K) {
    constexpr int BM = 128, BN = 64, BK = 32;
    constexpr int LDA = BK + 8;  // 40 halfs = 80 B rows
    constexpr int LDB = BK + 8;
    __shared__ _Float16 As[BM][LDA];
    __shared__ _Float16 BsT[BN][LDB];  // BsT[n][k] = W[n0+n][k0+k]

    const int t    = threadIdx.x;
    const int lane = t & 31;
    const int wave = t >> 5;
    const int wm   = wave >> 1;  // 0..3
    const int wn   = wave & 1;   // 0..1
    const int m0   = blockIdx.y * BM;
    const int n0   = blockIdx.x * BN;

    v8f acc[2][2];
#pragma unroll
    for (int i = 0; i < 2; ++i)
#pragma unroll
        for (int j = 0; j < 2; ++j)
#pragma unroll
            for (int e = 0; e < 8; ++e) acc[i][j][e] = 0.0f;

    for (int k0 = 0; k0 < K; k0 += BK) {
        // ---- Stage A tile. ----
        if constexpr (A_F16) {
            // 128x32 halfs = 512 v8h chunks; pure copy -> async DMA to LDS.
#pragma unroll
            for (int i = t; i < BM * BK / 8; i += 256) {
                const int r  = i >> 2;
                const int c8 = i & 3;
                copy_b128_to_lds((const _Float16*)Ap + (size_t)(m0 + r) * K + k0 + c8 * 8,
                                 &As[r][c8 * 8]);
            }
        } else {
            // 128x32 floats = 1024 float4 chunks; convert f32 -> f16.
#pragma unroll
            for (int i = t; i < BM * BK / 4; i += 256) {
                const int r  = i >> 3;
                const int c4 = i & 7;
                const float4 v = *(const float4*)((const float*)Ap +
                                                  (size_t)(m0 + r) * K + k0 + c4 * 4);
                *(v4h*)(&As[r][c4 * 4]) = cvt4(v);
            }
        }
        // ---- Stage W tile as-is (row-major copy): 64x32 floats = 512 float4. ----
#pragma unroll
        for (int i = t; i < BN * BK / 4; i += 256) {
            const int n  = i >> 3;
            const int c4 = i & 7;
            const float4 v = *(const float4*)(W + (size_t)(n0 + n) * K + k0 + c4 * 4);
            *(v4h*)(&BsT[n][c4 * 4]) = cvt4(v);
        }
        if constexpr (A_F16) wait_async_lds();
        __syncthreads();

        const v16h a0 = load_a_frag(&As[0][0], wm * 32, 0, LDA, lane);
        const v16h a1 = load_a_frag(&As[0][0], wm * 32 + 16, 0, LDA, lane);
        const v16h b0 = load_b_frag_T(&BsT[0][0], 0, wn * 32, LDB, lane);
        const v16h b1 = load_b_frag_T(&BsT[0][0], 0, wn * 32 + 16, LDB, lane);
        acc[0][0] = WMMA16x16x32(a0, b0, acc[0][0]);
        acc[0][1] = WMMA16x16x32(a0, b1, acc[0][1]);
        acc[1][0] = WMMA16x16x32(a1, b0, acc[1][0]);
        acc[1][1] = WMMA16x16x32(a1, b1, acc[1][1]);
        __syncthreads();
    }

    // Epilogue: bias (+ReLU); C/D layout: element j -> row j + 8*(lane>=16),
    // col = lane%16.
    const int hlf = lane >> 4;
    const int c16 = lane & 15;
#pragma unroll
    for (int i = 0; i < 2; ++i)
#pragma unroll
        for (int jf = 0; jf < 2; ++jf) {
            const int col = n0 + wn * 32 + jf * 16 + c16;
            const float bv = bias[col];
#pragma unroll
            for (int j = 0; j < 8; ++j) {
                const int row = m0 + wm * 32 + i * 16 + j + 8 * hlf;
                float v = acc[i][jf][j] + bv;
                if constexpr (RELU) v = fmaxf(v, 0.0f);
                if constexpr (OUT_F16)
                    ((_Float16*)Cp)[(size_t)row * N + col] = (_Float16)v;
                else
                    ((float*)Cp)[(size_t)row * N + col] = v;
            }
        }
}

// ---------------------------------------------------------------------------
// Flash-style attention: one block per (b, h, 64-row query tile).
// S = Q K^T / 8 (WMMA), key-padding + causal masks, online softmax, O += P V.
// Q and K tiles stream via async global->LDS DMA; V is loaded b128 and
// transposed into VsT ([d][key]) so every fragment gather is contiguous.
// ---------------------------------------------------------------------------
__global__ __launch_bounds__(256) void attention_kernel(const _Float16* __restrict__ Q,
                                                        const _Float16* __restrict__ Kf,
                                                        const _Float16* __restrict__ V,
                                                        _Float16* __restrict__ O,
                                                        const int* __restrict__ attn_mask) {
    __shared__ _Float16 Qs[64][72];    // [q][d]
    __shared__ _Float16 Ks[64][72];    // [key][d]
    __shared__ _Float16 VsT[64][72];   // [d][key]   (transposed V tile)
    __shared__ _Float16 Ps[64][72];    // [q][key]
    __shared__ float    Ss[64][64];
    __shared__ float    m_s[64], l_s[64], alpha_s[64], ksum[64];

    const int t    = threadIdx.x;
    const int lane = t & 31;
    const int wave = t >> 5;
    const int wm   = wave >> 1;  // 0..3 (rows of 16)
    const int wn   = wave & 1;   // 0..1 (cols of 32)
    const int qt   = blockIdx.x;
    const int bh   = blockIdx.y;
    const int b    = bh >> 4;    // H = 16
    const int h    = bh & 15;
    const int q0   = qt * 64;
    const size_t rowbase = (size_t)b * Lseq;
    const int cbase      = h * DH;
    const bool causal    = (attn_mask[0] != 0);

    // Query tile: 512 v8h chunks, pure copy -> async DMA straight into LDS.
#pragma unroll
    for (int i = t; i < 64 * 64 / 8; i += 256) {
        const int r  = i >> 3;
        const int c8 = i & 7;
        copy_b128_to_lds(Q + (rowbase + q0 + r) * Dm + cbase + c8 * 8,
                         &Qs[r][c8 * 8]);
    }
    if (t < 64) { m_s[t] = -3.0e38f; l_s[t] = 0.0f; }

    v8f o[2];
#pragma unroll
    for (int f = 0; f < 2; ++f)
#pragma unroll
        for (int e = 0; e < 8; ++e) o[f][e] = 0.0f;

    const int ktEnd = causal ? qt : (Lseq / 64 - 1);
    for (int kt = 0; kt <= ktEnd; ++kt) {
        __syncthreads();  // previous-iter LDS consumers done
        const int k0g = kt * 64;
        // K tile: async DMA copy; V tile: b128 load + transposed scatter.
#pragma unroll
        for (int i = t; i < 64 * 64 / 8; i += 256) {
            const int r  = i >> 3;   // key index
            const int c8 = i & 7;    // d chunk
            const size_t g = (rowbase + k0g + r) * Dm + cbase + c8 * 8;
            copy_b128_to_lds(Kf + g, &Ks[r][c8 * 8]);
            const v8h vv = *(const v8h*)(V + g);
#pragma unroll
            for (int e = 0; e < 8; ++e) VsT[c8 * 8 + e][r] = vv[e];
        }
        wait_async_lds();  // covers Q (iter 0) and K asyncs of this wave
        __syncthreads();

        // Key-padding mask precursor: per-key sum over head dim.
        if (t < 64) {
            float s = 0.0f;
#pragma unroll
            for (int d = 0; d < 64; ++d) s += (float)Ks[t][d];
            ksum[t] = s;
        }

        // S = Q * K^T (contraction over d = 64 -> two k=32 WMMA steps).
        v8f sa[2];
#pragma unroll
        for (int f = 0; f < 2; ++f)
#pragma unroll
            for (int e = 0; e < 8; ++e) sa[f][e] = 0.0f;
#pragma unroll
        for (int kd = 0; kd < 64; kd += 32) {
            const v16h aq  = load_a_frag(&Qs[0][0], wm * 16, kd, 72, lane);
            const v16h bk0 = load_b_frag_T(&Ks[0][0], kd, wn * 32, 72, lane);
            const v16h bk1 = load_b_frag_T(&Ks[0][0], kd, wn * 32 + 16, 72, lane);
            sa[0] = WMMA16x16x32(aq, bk0, sa[0]);
            sa[1] = WMMA16x16x32(aq, bk1, sa[1]);
        }
        {
            const int hlf = lane >> 4, c16 = lane & 15;
#pragma unroll
            for (int f = 0; f < 2; ++f)
#pragma unroll
                for (int j = 0; j < 8; ++j)
                    Ss[wm * 16 + j + 8 * hlf][wn * 32 + f * 16 + c16] =
                        sa[f][j] * 0.125f;  // 1/sqrt(dh), dh=64
        }
        __syncthreads();

        // Online softmax row update (thread t owns query row t).
        if (t < 64) {
            const int qg = q0 + t;
            const float mold = m_s[t];
            float mnew = mold;
#pragma unroll 8
            for (int j = 0; j < 64; ++j) {
                float s = Ss[t][j];
                const bool masked = (ksum[j] == 0.0f) || (causal && (k0g + j > qg));
                s = masked ? NEG_MASK : s;
                Ss[t][j] = s;
                mnew = fmaxf(mnew, s);
            }
            const float alpha = __expf(mold - mnew);
            float lsum = 0.0f;
#pragma unroll 8
            for (int j = 0; j < 64; ++j) {
                const float p = __expf(Ss[t][j] - mnew);
                Ps[t][j] = (_Float16)p;
                lsum += p;
            }
            m_s[t]     = mnew;
            l_s[t]     = l_s[t] * alpha + lsum;
            alpha_s[t] = alpha;
        }
        __syncthreads();

        // Rescale running O by alpha, then O += P * V.
        {
            const int hlf = lane >> 4;
#pragma unroll
            for (int f = 0; f < 2; ++f)
#pragma unroll
                for (int j = 0; j < 8; ++j)
                    o[f][j] *= alpha_s[wm * 16 + j + 8 * hlf];
        }
#pragma unroll
        for (int kd = 0; kd < 64; kd += 32) {
            const v16h ap  = load_a_frag(&Ps[0][0], wm * 16, kd, 72, lane);
            // B[k=key][n=d] = VsT[d][key]: contiguous per-lane gather.
            const v16h bv0 = load_b_frag_T(&VsT[0][0], kd, wn * 32, 72, lane);
            const v16h bv1 = load_b_frag_T(&VsT[0][0], kd, wn * 32 + 16, 72, lane);
            o[0] = WMMA16x16x32(ap, bv0, o[0]);
            o[1] = WMMA16x16x32(ap, bv1, o[1]);
        }
    }
    __syncthreads();

    // Finalize: O /= l, store f16 to [B*L, D] at this head's column slice.
    {
        const int hlf = lane >> 4, c16 = lane & 15;
#pragma unroll
        for (int f = 0; f < 2; ++f)
#pragma unroll
            for (int j = 0; j < 8; ++j) {
                const int r = wm * 16 + j + 8 * hlf;
                const float v = o[f][j] / l_s[r];
                O[(rowbase + q0 + r) * Dm + cbase + wn * 32 + f * 16 + c16] =
                    (_Float16)v;
            }
    }
}

// ---------------------------------------------------------------------------
// Host launcher.
// Inputs: query,key,value,Wq,bq,Wk,bk,Wv,bv,Wo,bo,attn_mask (fp32 / int)
// ---------------------------------------------------------------------------
extern "C" void kernel_launch(void* const* d_in, const int* in_sizes, int n_in,
                              void* d_out, int out_size, void* d_ws, size_t ws_size,
                              hipStream_t stream) {
    const float* query = (const float*)d_in[0];
    const float* key   = (const float*)d_in[1];
    const float* value = (const float*)d_in[2];
    const float* Wq = (const float*)d_in[3];
    const float* bq = (const float*)d_in[4];
    const float* Wk = (const float*)d_in[5];
    const float* bk = (const float*)d_in[6];
    const float* Wv = (const float*)d_in[7];
    const float* bv = (const float*)d_in[8];
    const float* Wo = (const float*)d_in[9];
    const float* bo = (const float*)d_in[10];
    const int* attn_mask = (const int*)d_in[11];

    const size_t mat = (size_t)Mrows * Dm;  // 4M elements
    _Float16* Qh = (_Float16*)d_ws;         // 8 MB each, 32 MB total
    _Float16* Kh = Qh + mat;
    _Float16* Vh = Kh + mat;
    _Float16* Ah = Vh + mat;

    const dim3 gg(Dm / 64, Mrows / 128);  // (16, 32)
    gemm_bias_act<false, true, true><<<gg, 256, 0, stream>>>(query, Wq, bq, Qh, Mrows, Dm, Dm);
    gemm_bias_act<false, true, true><<<gg, 256, 0, stream>>>(key,   Wk, bk, Kh, Mrows, Dm, Dm);
    gemm_bias_act<false, true, true><<<gg, 256, 0, stream>>>(value, Wv, bv, Vh, Mrows, Dm, Dm);

    attention_kernel<<<dim3(Lseq / 64, Bsz * Hh), 256, 0, stream>>>(Qh, Kh, Vh, Ah, attn_mask);

    gemm_bias_act<true, false, false><<<gg, 256, 0, stream>>>(Ah, Wo, bo, (float*)d_out, Mrows, Dm, Dm);
}
